// AttentiveGraphNet_85186381349285
// MI455X (gfx1250) — matrix operand order repaired and checked by reading, not compiled, not run on hardware
//
#include <hip/hip_runtime.h>
#include <hip/hip_fp16.h>

typedef __attribute__((ext_vector_type(16))) _Float16 v16h;
typedef __attribute__((ext_vector_type(8)))  float    v8f;

namespace {
constexpr int    kN     = 36864;
constexpr int    kHid   = 192;
constexpr int    kHeads = 6;
constexpr int    kDh    = 32;
constexpr int    kE     = 294912;
constexpr int    kNG    = 96;
constexpr int    kLG    = 384;
constexpr int    kM     = 64;          // performer features
constexpr size_t kNB    = (size_t)kN * kHid;   // 7,077,888 floats
constexpr int    kEH    = kE * kHeads;         // 1,769,472
constexpr float  kInvSqrtDh = 0.17677669529663689f;
constexpr float  kS4        = 0.4204482076268573f;  // 32^-0.25
}

// ---------------------------------------------------------------- helpers
static __device__ __forceinline__ float warp_sum(float v) {
#pragma unroll
  for (int off = 16; off >= 1; off >>= 1) v += __shfl_xor(v, off, 32);
  return v;
}
static __device__ __forceinline__ unsigned ford(float f) {  // order-preserving f32->u32
  unsigned u = __float_as_uint(f);
  return (u & 0x80000000u) ? ~u : (u | 0x80000000u);
}
static __device__ __forceinline__ float fdec(unsigned u) {
  unsigned b = (u & 0x80000000u) ? (u ^ 0x80000000u) : ~u;
  return __uint_as_float(b);
}

// ---------------------------------------------------------------- WMMA GEMM
// C[M,Nc] = A[M,Kd] @ W[Kd,Nc] + bias (+gelu | +res).  64x64 block tile,
// 8 waves x two 16x16 WMMA tiles, f16 operands staged in LDS, f32 accum.
template <int EPI>  // 0 = bias, 1 = bias+gelu(erf), 2 = bias+residual
__global__ void __launch_bounds__(256) gemm_wmma(
    const float* __restrict__ A, const float* __restrict__ W,
    const float* __restrict__ bias, const float* __restrict__ res,
    float* __restrict__ C, int Kd, int Nc) {
  __shared__ _Float16 aS[64 * 32];   // [row][k]
  __shared__ _Float16 bS[64 * 32];   // [col][k] (transposed stage)
  const int tid = threadIdx.x, lane = tid & 31, wave = tid >> 5;
  const int hi = lane >> 4, ln = lane & 15;
  const int row0 = blockIdx.y * 64, col0 = blockIdx.x * 64;
  const int rs = wave & 3, cs = (wave >> 2) * 2;
  v8f acc0 = {}, acc1 = {};
  for (int k0 = 0; k0 < Kd; k0 += 32) {
    if (k0 + 32 < Kd)  // prefetch next A K-slab (global_prefetch_b8)
      __builtin_prefetch(&A[(size_t)(row0 + (tid >> 5)) * Kd + k0 + 32 + (tid & 31)], 0, 1);
#pragma unroll
    for (int i = 0; i < 8; ++i) {
      int idx = tid + i * 256;                 // 64x32 elements
      int r = idx >> 5, kk = idx & 31;
      aS[idx] = (_Float16)A[(size_t)(row0 + r) * Kd + (k0 + kk)];
    }
#pragma unroll
    for (int i = 0; i < 8; ++i) {
      int idx = tid + i * 256;                 // 32x64 elements
      int kk = idx >> 6, c = idx & 63;
      bS[c * 32 + kk] = (_Float16)W[(size_t)(k0 + kk) * Nc + (col0 + c)];
    }
    __syncthreads();
    union { v16h v; unsigned u[8]; } af, bf0, bf1;
    const unsigned* aU = (const unsigned*)aS;
    const unsigned* bU = (const unsigned*)bS;
    const int arow = rs * 16 + ln;
    const int bc0 = cs * 16 + ln, bc1 = bc0 + 16;
#pragma unroll
    for (int j = 0; j < 8; ++j) {
      int kb = ((j < 4) ? (2 * j) : (2 * j + 8)) + hi * 8;  // A-frag K pattern
      af.u[j] = aU[arow * 16 + (kb >> 1)];
      int kbb = ((j < 4) ? (2 * j) : (2 * j + 8)) + hi * 8;  // B-frag K pattern
      bf0.u[j] = bU[bc0 * 16 + (kbb >> 1)];
      bf1.u[j] = bU[bc1 * 16 + (kbb >> 1)];
    }
    acc0 = __builtin_amdgcn_wmma_f32_16x16x32_f16(false, af.v, false, bf0.v, (short)0, acc0, false, false);
    acc1 = __builtin_amdgcn_wmma_f32_16x16x32_f16(false, af.v, false, bf1.v, (short)0, acc1, false, false);
    __syncthreads();
  }
#pragma unroll
  for (int r = 0; r < 8; ++r) {
    size_t m = (size_t)row0 + rs * 16 + r + 8 * hi;
    int c0 = col0 + cs * 16 + ln, c1 = c0 + 16;
    float v0 = acc0[r] + bias[c0];
    float v1 = acc1[r] + bias[c1];
    if (EPI == 1) {
      v0 = 0.5f * v0 * (1.f + erff(v0 * 0.7071067811865476f));
      v1 = 0.5f * v1 * (1.f + erff(v1 * 0.7071067811865476f));
    }
    if (EPI == 2) { v0 += res[m * Nc + c0]; v1 += res[m * Nc + c1]; }
    C[m * Nc + c0] = v0;
    C[m * Nc + c1] = v1;
  }
}

// ---------------------------------------------------------------- fused edge encoder
// wc[3*192] = edge_enc.w @ e.w ; bc[192] = edge_enc.b @ e.w  (per GPS layer)
__global__ void __launch_bounds__(192) build_wcomb(
    const float* __restrict__ w1, const float* __restrict__ b1,
    const float* __restrict__ ew, float* __restrict__ out) {
  int c = threadIdx.x;
  float s0 = 0.f, s1 = 0.f, s2 = 0.f, sb = 0.f;
  for (int j = 0; j < 192; ++j) {
    float e = ew[(size_t)j * 192 + c];
    s0 += w1[j] * e; s1 += w1[192 + j] * e; s2 += w1[384 + j] * e; sb += b1[j] * e;
  }
  out[c] = s0; out[192 + c] = s1; out[384 + c] = s2; out[576 + c] = sb;
}

// ---------------------------------------------------------------- TransformerConv edge passes
__global__ void __launch_bounds__(256) edge_logits(
    const int* __restrict__ src, const int* __restrict__ dst,
    const float* __restrict__ ea, const float* __restrict__ q,
    const float* __restrict__ k, const float* __restrict__ wcb,
    float* __restrict__ elog, unsigned* __restrict__ lmax) {
  __shared__ float wS[768];
  for (int i = threadIdx.x; i < 768; i += 256) wS[i] = wcb[i];
  __syncthreads();
  int idx = blockIdx.x * 256 + threadIdx.x;
  if (idx >= kEH) return;
  int e = idx / kHeads, h = idx % kHeads;
  int s = src[e], d = dst[e];
  float a0 = ea[e * 3 + 0], a1 = ea[e * 3 + 1], a2 = ea[e * 3 + 2];
  const float* qr = q + (size_t)d * kHid + h * kDh;
  const float* kr = k + (size_t)s * kHid + h * kDh;
  float lg = 0.f;
#pragma unroll
  for (int t = 0; t < kDh; ++t) {
    int c = h * kDh + t;
    float ev = wS[576 + c] + a0 * wS[c] + a1 * wS[192 + c] + a2 * wS[384 + c];
    lg += qr[t] * (kr[t] + ev);
  }
  lg *= kInvSqrtDh;
  elog[idx] = lg;
  atomicMax(&lmax[(size_t)d * kHeads + h], ford(lg));
}

__global__ void __launch_bounds__(256) edge_exp(
    const int* __restrict__ dst, float* __restrict__ elog,
    const unsigned* __restrict__ lmax, float* __restrict__ den) {
  int idx = blockIdx.x * 256 + threadIdx.x;
  if (idx >= kEH) return;
  int e = idx / kHeads, h = idx % kHeads;
  int d = dst[e];
  float mx = fdec(lmax[(size_t)d * kHeads + h]);
  if (!(mx > -1e30f && mx < 1e30f)) mx = 0.f;   // reference: non-finite lmax -> 0
  float ex = __expf(elog[idx] - mx);
  elog[idx] = ex;
  atomicAdd(&den[(size_t)d * kHeads + h], ex);
}

__global__ void __launch_bounds__(256) edge_agg(
    const int* __restrict__ src, const int* __restrict__ dst,
    const float* __restrict__ ea, const float* __restrict__ v,
    const float* __restrict__ wcb, const float* __restrict__ elog,
    const float* __restrict__ den, float* __restrict__ out) {
  __shared__ float wS[768];
  for (int i = threadIdx.x; i < 768; i += 256) wS[i] = wcb[i];
  __syncthreads();
  int idx = blockIdx.x * 256 + threadIdx.x;
  if (idx >= kEH) return;
  int e = idx / kHeads, h = idx % kHeads;
  int s = src[e], d = dst[e];
  float a0 = ea[e * 3 + 0], a1 = ea[e * 3 + 1], a2 = ea[e * 3 + 2];
  float alpha = elog[idx] / (den[(size_t)d * kHeads + h] + 1e-16f);
  const float* vr = v + (size_t)s * kHid + h * kDh;
  float* orow = out + (size_t)d * kHid;
#pragma unroll
  for (int t = 0; t < kDh; ++t) {
    int c = h * kDh + t;
    float ev = wS[576 + c] + a0 * wS[c] + a1 * wS[192 + c] + a2 * wS[384 + c];
    atomicAdd(&orow[c], (vr[t] + ev) * alpha);
  }
}

// beta-gated skip: out = beta*xr + (1-beta)*conv + h   (one wave per node)
__global__ void __launch_bounds__(256) beta_gate(
    const float* __restrict__ o, const float* __restrict__ x,
    const float* __restrict__ hin, const float* __restrict__ bw,
    float* __restrict__ out) {
  int node = blockIdx.x * 8 + (threadIdx.x >> 5);
  int lane = threadIdx.x & 31;
  const float* orow = o + (size_t)node * kHid;
  const float* xrow = x + (size_t)node * kHid;
  const float* hrow = hin + (size_t)node * kHid;
  float dot = 0.f;
  for (int i = lane; i < kHid; i += 32) {
    float ov = orow[i], xv = xrow[i];
    dot += ov * bw[i] + xv * bw[kHid + i] + (ov - xv) * bw[2 * kHid + i];
  }
  dot = warp_sum(dot);
  float bg = 1.f / (1.f + __expf(-dot));
  for (int i = lane; i < kHid; i += 32)
    out[(size_t)node * kHid + i] = bg * xrow[i] + (1.f - bg) * orow[i] + hrow[i];
}

// ---------------------------------------------------------------- flash MHA (WMMA)
// Row sums of P are computed with an extra WMMA against an all-ones B tile
// (replaces a 4-step ds_bpermute tree per row); only row-max uses shuffles.
__global__ void __launch_bounds__(256) mha_flash(
    const float* __restrict__ Q, const float* __restrict__ K,
    const float* __restrict__ V, float* __restrict__ O) {
  __shared__ _Float16 kS[kLG * kDh];      // [l][d]
  __shared__ _Float16 vS[kDh * kLG];      // [d][l]  (transposed)
  __shared__ _Float16 pS[8 * 16 * 32];    // per-wave P tile
  int gh = blockIdx.x;
  int g = gh / kHeads, h = gh % kHeads;
  size_t base = (size_t)g * kLG * kHid + (size_t)h * kDh;
  int tid = threadIdx.x, lane = tid & 31, wave = tid >> 5;
  int hi = lane >> 4, ln = lane & 15;
  for (int i = tid; i < kLG * kDh; i += 256) {
    int l = i >> 5, d = i & 31;
    kS[i] = (_Float16)K[base + (size_t)l * kHid + d];
    vS[d * kLG + l] = (_Float16)V[base + (size_t)l * kHid + d];
  }
  __syncthreads();
  v16h onesf;
#pragma unroll
  for (int j = 0; j < 16; ++j) onesf[j] = (_Float16)1.f;
  _Float16* myP = pS + wave * 512;
  for (int qt = wave; qt < kLG / 16; qt += 8) {       // 3 query tiles per wave
    v16h qf;
    {
      const float* qr = Q + base + (size_t)(qt * 16 + ln) * kHid;
#pragma unroll
      for (int j = 0; j < 8; ++j) {
        int kb = ((j < 4) ? (2 * j) : (2 * j + 8)) + hi * 8;
        qf[2 * j] = (_Float16)(qr[kb] * kInvSqrtDh);
        qf[2 * j + 1] = (_Float16)(qr[kb + 1] * kInvSqrtDh);
      }
    }
    float mrow[8], lrow[8], sc[8];
#pragma unroll
    for (int r = 0; r < 8; ++r) { mrow[r] = -1e30f; lrow[r] = 0.f; }
    v8f a0 = {}, a1 = {};
    for (int kt = 0; kt < kLG / 32; ++kt) {
      int kbase = kt * 32;
      v16h b0, b1;
#pragma unroll
      for (int j = 0; j < 8; ++j) {
        int d = ((j < 4) ? (2 * j) : (2 * j + 8)) + hi * 8;
        int n0 = kbase + ln, n1 = kbase + 16 + ln;
        b0[2 * j] = kS[n0 * 32 + d];     b0[2 * j + 1] = kS[n0 * 32 + d + 1];
        b1[2 * j] = kS[n1 * 32 + d];     b1[2 * j + 1] = kS[n1 * 32 + d + 1];
      }
      v8f zc = {};
      v8f s0 = __builtin_amdgcn_wmma_f32_16x16x32_f16(false, qf, false, b0, (short)0, zc, false, false);
      v8f s1 = __builtin_amdgcn_wmma_f32_16x16x32_f16(false, qf, false, b1, (short)0, zc, false, false);
#pragma unroll
      for (int r = 0; r < 8; ++r) {       // online max per row (M = r + 8*hi)
        float mx = fmaxf(s0[r], s1[r]);
#pragma unroll
        for (int off = 8; off >= 1; off >>= 1) mx = fmaxf(mx, __shfl_xor(mx, off, 32));
        float mnew = fmaxf(mrow[r], mx);
        sc[r] = __expf(mrow[r] - mnew);
        mrow[r] = mnew;
        float p0 = __expf(s0[r] - mnew);
        float p1 = __expf(s1[r] - mnew);
        a0[r] = a0[r] * sc[r];
        a1[r] = a1[r] * sc[r];
        myP[(r + 8 * hi) * 32 + ln] = (_Float16)p0;
        myP[(r + 8 * hi) * 32 + 16 + ln] = (_Float16)p1;
      }
      // per-wave P region: same-wave LDS ops are in-order (ISA ch.11), the
      // backend inserts s_wait_dscnt on the dependent loads -- no barrier.
      v16h pf, vb0, vb1;
#pragma unroll
      for (int j = 0; j < 8; ++j) {
        int kb = ((j < 4) ? (2 * j) : (2 * j + 8)) + hi * 8;
        pf[2 * j] = myP[ln * 32 + kb];  pf[2 * j + 1] = myP[ln * 32 + kb + 1];
        int kk = ((j < 4) ? (2 * j) : (2 * j + 8)) + hi * 8;
        vb0[2 * j] = vS[ln * kLG + kbase + kk];
        vb0[2 * j + 1] = vS[ln * kLG + kbase + kk + 1];
        vb1[2 * j] = vS[(16 + ln) * kLG + kbase + kk];
        vb1[2 * j + 1] = vS[(16 + ln) * kLG + kbase + kk + 1];
      }
      v8f zs = {};
      v8f st = __builtin_amdgcn_wmma_f32_16x16x32_f16(false, pf, false, onesf, (short)0, zs, false, false);
#pragma unroll
      for (int r = 0; r < 8; ++r) lrow[r] = lrow[r] * sc[r] + st[r];  // row sums via WMMA
      a0 = __builtin_amdgcn_wmma_f32_16x16x32_f16(false, pf, false, vb0, (short)0, a0, false, false);
      a1 = __builtin_amdgcn_wmma_f32_16x16x32_f16(false, pf, false, vb1, (short)0, a1, false, false);
    }
#pragma unroll
    for (int r = 0; r < 8; ++r) {
      float inv = 1.f / lrow[r];
      size_t m = qt * 16 + r + 8 * hi;
      O[base + m * kHid + ln] = a0[r] * inv;
      O[base + m * kHid + 16 + ln] = a1[r] * inv;
    }
  }
}

// ---------------------------------------------------------------- Performer (FAVOR+)
__global__ void __launch_bounds__(256) performer_attn(
    const float* __restrict__ Q, const float* __restrict__ K,
    const float* __restrict__ V, const float* __restrict__ proj,
    float* __restrict__ O) {
  __shared__ float projS[kM * kDh];
  __shared__ float kvS[kM * kDh];
  __shared__ float ksumS[kM];
  __shared__ float kpS[128 * kM];
  int gh = blockIdx.x;
  int g = gh / kHeads, h = gh % kHeads;
  size_t base = (size_t)g * kLG * kHid + (size_t)h * kDh;
  int tid = threadIdx.x;
  for (int i = tid; i < kM * kDh; i += 256) { projS[i] = proj[i]; kvS[i] = 0.f; }
  if (tid < kM) ksumS[tid] = 0.f;
  __syncthreads();
  for (int chunk = 0; chunk < 3; ++chunk) {
    int l0 = chunk * 128;
    for (int s = tid; s < 128 * kM; s += 256) {
      int l = s >> 6, m = s & 63;
      const float* kr = K + base + (size_t)(l0 + l) * kHid;
      float pr = 0.f, sq = 0.f;
#pragma unroll
      for (int d = 0; d < kDh; ++d) { float t = kr[d] * kS4; pr += t * projS[m * kDh + d]; sq += t * t; }
      kpS[s] = __expf(pr - 0.5f * sq) * 0.125f;   // 1/sqrt(M)
    }
    __syncthreads();
    for (int s = tid; s < kM * kDh; s += 256) {
      int m = s >> 5, d = s & 31;
      float a = 0.f;
      for (int l = 0; l < 128; ++l)
        a += kpS[l * kM + m] * V[base + (size_t)(l0 + l) * kHid + d];
      kvS[s] += a;
      if (d == 0) {
        float ks = 0.f;
        for (int l = 0; l < 128; ++l) ks += kpS[l * kM + m];
        ksumS[m] += ks;
      }
    }
    __syncthreads();
  }
  for (int l = tid; l < kLG; l += 256) {
    const float* qr = Q + base + (size_t)l * kHid;
    float qv[kDh]; float sq = 0.f;
#pragma unroll
    for (int d = 0; d < kDh; ++d) { float t = qr[d] * kS4; qv[d] = t; sq += t * t; }
    float num[kDh];
#pragma unroll
    for (int d = 0; d < kDh; ++d) num[d] = 0.f;
    float den = 0.f;
    for (int m = 0; m < kM; ++m) {
      float pr = 0.f;
#pragma unroll
      for (int d = 0; d < kDh; ++d) pr += qv[d] * projS[m * kDh + d];
      float qp = __expf(pr - 0.5f * sq) * 0.125f;
      den += qp * ksumS[m];
#pragma unroll
      for (int d = 0; d < kDh; ++d) num[d] += qp * kvS[m * kDh + d];
    }
    float z = 1.f / (den + 1e-6f);
#pragma unroll
    for (int d = 0; d < kDh; ++d) O[base + (size_t)l * kHid + d] = num[d] * z;
  }
}

// ---------------------------------------------------------------- norms / elementwise
__global__ void __launch_bounds__(256) gln_stats(const float* __restrict__ x, float* __restrict__ st) {
  const int LGF = kLG * kHid;
  int g = blockIdx.x;
  const float* p = x + (size_t)g * LGF;
  float s = 0.f, ss = 0.f;
  for (int i = threadIdx.x; i < LGF; i += 256) { float v = p[i]; s += v; ss += v * v; }
  s = warp_sum(s); ss = warp_sum(ss);
  __shared__ float rs[8], rss[8];
  int wave = threadIdx.x >> 5, lane = threadIdx.x & 31;
  if (lane == 0) { rs[wave] = s; rss[wave] = ss; }
  __syncthreads();
  if (threadIdx.x == 0) {
    float S = 0.f, SS = 0.f;
    for (int w = 0; w < 8; ++w) { S += rs[w]; SS += rss[w]; }
    float mean = S / LGF;
    float var = SS / LGF - mean * mean;
    st[2 * g] = mean;
    st[2 * g + 1] = rsqrtf(var + 1e-5f);
  }
}
__global__ void __launch_bounds__(256) gln_apply(
    const float* __restrict__ x, const float* __restrict__ st,
    const float* __restrict__ w, const float* __restrict__ b,
    float* __restrict__ y, size_t total) {
  for (size_t i = (size_t)blockIdx.x * 256 + threadIdx.x; i < total; i += (size_t)gridDim.x * 256) {
    int c = (int)(i % kHid);
    int g = (int)(i / ((size_t)kHid * kLG));
    y[i] = (x[i] - st[2 * g]) * st[2 * g + 1] * w[c] + b[c];
  }
}
__global__ void __launch_bounds__(256) glob_stats(const float* __restrict__ x, float* __restrict__ acc, size_t total) {
  float s = 0.f, ss = 0.f;
  for (size_t i = (size_t)blockIdx.x * 256 + threadIdx.x; i < total; i += (size_t)gridDim.x * 256) {
    float v = x[i]; s += v; ss += v * v;
  }
  s = warp_sum(s); ss = warp_sum(ss);
  __shared__ float rs[8], rss[8];
  int wave = threadIdx.x >> 5, lane = threadIdx.x & 31;
  if (lane == 0) { rs[wave] = s; rss[wave] = ss; }
  __syncthreads();
  if (threadIdx.x == 0) {
    float S = 0.f, SS = 0.f;
    for (int w = 0; w < 8; ++w) { S += rs[w]; SS += rss[w]; }
    atomicAdd(&acc[0], S);
    atomicAdd(&acc[1], SS);
  }
}
__global__ void __launch_bounds__(256) glob_apply(
    const float* __restrict__ x, const float* __restrict__ acc,
    const float* __restrict__ w, const float* __restrict__ b,
    float* __restrict__ y, size_t total) {
  float mean = acc[0] / (float)total;
  float var = acc[1] / (float)total - mean * mean;
  float inv = rsqrtf(var + 1e-5f);
  for (size_t i = (size_t)blockIdx.x * 256 + threadIdx.x; i < total; i += (size_t)gridDim.x * 256) {
    int c = (int)(i % kHid);
    y[i] = (x[i] - mean) * inv * w[c] + b[c];
  }
}
__global__ void __launch_bounds__(256) add2(const float* __restrict__ a, const float* __restrict__ b,
                                            float* __restrict__ y, size_t total) {
  for (size_t i = (size_t)blockIdx.x * 256 + threadIdx.x; i < total; i += (size_t)gridDim.x * 256)
    y[i] = a[i] + b[i];
}

// ---------------------------------------------------------------- embed / output
__global__ void __launch_bounds__(256) embed_kernel(
    const int* __restrict__ tok, const float* __restrict__ pe,
    const float* __restrict__ emb, const float* __restrict__ pnb,
    const float* __restrict__ pnw, const float* __restrict__ peb,
    const float* __restrict__ pew, float* __restrict__ h) {
  __shared__ float wS[20 * 64];
  for (int i = threadIdx.x; i < 1280; i += 256) wS[i] = pew[i];
  __syncthreads();
  int n = blockIdx.x * 256 + threadIdx.x;
  if (n >= kN) return;
  const float* p = pe + (size_t)n * 20;
  float mu = 0.f;
#pragma unroll
  for (int i = 0; i < 20; ++i) mu += p[i];
  mu *= (1.f / 20.f);
  float var = 0.f; float t[20];
#pragma unroll
  for (int i = 0; i < 20; ++i) { float d = p[i] - mu; t[i] = d; var += d * d; }
  var *= (1.f / 20.f);
  float inv = rsqrtf(var + 1e-5f);
#pragma unroll
  for (int i = 0; i < 20; ++i) t[i] = t[i] * inv * pnw[i] + pnb[i];
  int tk = tok[n];
  float* hr = h + (size_t)n * kHid;
  const float* er = emb + (size_t)tk * 128;
  for (int c = 0; c < 128; ++c) hr[c] = er[c];
  for (int c = 0; c < 64; ++c) {
    float a = peb[c];
#pragma unroll
    for (int i = 0; i < 20; ++i) a += t[i] * wS[i * 64 + c];
    hr[128 + c] = a;
  }
}
__global__ void __launch_bounds__(256) out_proj(
    const float* __restrict__ h, const float* __restrict__ w,
    const float* __restrict__ b, float* __restrict__ o) {
  int n = blockIdx.x * 256 + threadIdx.x;
  if (n >= kN) return;
  const float* hr = h + (size_t)n * kHid;
  float s0 = b[0], s1 = b[1];
  for (int c = 0; c < kHid; ++c) { float v = hr[c]; s0 += v * w[2 * c]; s1 += v * w[2 * c + 1]; }
  o[2 * n] = s0; o[2 * n + 1] = s1;
}

// ---------------------------------------------------------------- host side
struct LinP { const float* b; const float* w; };
struct GpsP {
  const float* proj;
  LinP wk, wo, wq, wv;
  const float* beta_w; const float* e_w;
  LinP ck, cq, cskip, cv;
  LinP l1, l2;
  const float *n1b, *n1w, *n2b, *n2w, *n3b, *n3w;
};

extern "C" void kernel_launch(void* const* d_in, const int* in_sizes, int n_in,
                              void* d_out, int out_size, void* d_ws, size_t ws_size,
                              hipStream_t stream) {
  (void)in_sizes; (void)n_in; (void)out_size; (void)ws_size;
  const int*   xtok = (const int*)d_in[0];
  const int*   ei   = (const int*)d_in[1];
  const float* ea   = (const float*)d_in[2];
  const float* pe   = (const float*)d_in[4];
  const float* cur  = (const float*)d_in[5];

  auto take = [&](size_t n) { const float* r = cur; cur += n; return r; };
  auto takeLin = [&](size_t din, size_t dout) { LinP L; L.b = take(dout); L.w = take(din * dout); return L; };
  auto takeGps = [&](bool perf) {
    GpsP G{};
    if (perf) G.proj = take((size_t)kM * kDh);
    G.wk = takeLin(192, 192); G.wo = takeLin(192, 192); G.wq = takeLin(192, 192); G.wv = takeLin(192, 192);
    G.beta_w = take(576); G.e_w = take(192 * 192);
    G.ck = takeLin(192, 192); G.cq = takeLin(192, 192); G.cskip = takeLin(192, 192); G.cv = takeLin(192, 192);
    G.l1 = takeLin(192, 384); G.l2 = takeLin(384, 192);
    G.n1b = take(192); G.n1w = take(192);
    G.n2b = take(192); G.n2w = take(192);
    G.n3b = take(192); G.n3w = take(192);
    return G;
  };
  // jax tree-flatten order (sorted dict keys)
  LinP edge_enc = takeLin(3, 192);
  const float* node_emb = take(5 * 128);
  const float* norm_b = take(192); const float* norm_w = take(192);
  LinP outp = takeLin(192, 2);
  GpsP p0 = takeGps(true), p1 = takeGps(true);
  LinP pe_enc = takeLin(20, 64);
  const float* pe_norm_b = take(20); const float* pe_norm_w = take(20);
  GpsP t0 = takeGps(false), t1l = takeGps(false);

  float* ws = (float*)d_ws;
  float* h_    = ws;
  float* ht_   = ws + kNB;
  float* hp_   = ws + 2 * kNB;
  float* qb_   = ws + 3 * kNB;
  float* kb_   = ws + 4 * kNB;
  float* vb_   = ws + 5 * kNB;
  float* skip_ = ws + 6 * kNB;
  float* conv_ = ws + 7 * kNB;
  float* t1_   = ws + 8 * kNB;
  float* t2_   = ws + 9 * kNB;
  float* elog_ = ws + 10 * kNB;
  unsigned* lmax_ = (unsigned*)(elog_ + kEH);
  float* den_  = (float*)(lmax_ + (size_t)kN * kHeads);
  float* gst_  = den_ + (size_t)kN * kHeads;
  float* gacc_ = gst_ + 2 * kNG;
  float* wcomb_ = gacc_ + 8;
  float* mlph_ = qb_;  // [N,384] aliases qb_+kb_ (free by MLP stage)

  const int* src  = ei;
  const int* dsti = ei + kE;

  // fused 3x192 edge encoders (order: t0, p0, t1, p1)
  build_wcomb<<<1, 192, 0, stream>>>(edge_enc.w, edge_enc.b, t0.e_w,  wcomb_ + 0 * 768);
  build_wcomb<<<1, 192, 0, stream>>>(edge_enc.w, edge_enc.b, p0.e_w,  wcomb_ + 1 * 768);
  build_wcomb<<<1, 192, 0, stream>>>(edge_enc.w, edge_enc.b, t1l.e_w, wcomb_ + 2 * 768);
  build_wcomb<<<1, 192, 0, stream>>>(edge_enc.w, edge_enc.b, p1.e_w,  wcomb_ + 3 * 768);

  embed_kernel<<<kN / 256, 256, 0, stream>>>(xtok, pe, node_emb, pe_norm_b, pe_norm_w,
                                             pe_enc.b, pe_enc.w, h_);

  dim3 g192(3, kN / 64), g384(6, kN / 64);
  auto run_gps = [&](const GpsP& P, bool performer, const float* hin, float* dstb, const float* wc) {
    // local MPNN branch: q/k/v/skip projections
    gemm_wmma<0><<<g192, 256, 0, stream>>>(hin, P.cq.w,    P.cq.b,    nullptr, qb_,   kHid, kHid);
    gemm_wmma<0><<<g192, 256, 0, stream>>>(hin, P.ck.w,    P.ck.b,    nullptr, kb_,   kHid, kHid);
    gemm_wmma<0><<<g192, 256, 0, stream>>>(hin, P.cv.w,    P.cv.b,    nullptr, vb_,   kHid, kHid);
    gemm_wmma<0><<<g192, 256, 0, stream>>>(hin, P.cskip.w, P.cskip.b, nullptr, skip_, kHid, kHid);
    hipMemsetAsync(lmax_, 0, (size_t)kN * kHeads * 4, stream);
    hipMemsetAsync(den_,  0, (size_t)kN * kHeads * 4, stream);
    hipMemsetAsync(conv_, 0, kNB * 4, stream);
    edge_logits<<<kEH / 256, 256, 0, stream>>>(src, dsti, ea, qb_, kb_, wc, elog_, lmax_);
    edge_exp<<<kEH / 256, 256, 0, stream>>>(dsti, elog_, lmax_, den_);
    edge_agg<<<kEH / 256, 256, 0, stream>>>(src, dsti, ea, vb_, wc, elog_, den_, conv_);
    beta_gate<<<kN / 8, 256, 0, stream>>>(conv_, skip_, hin, P.beta_w, t1_);
    gln_stats<<<kNG, 256, 0, stream>>>(t1_, gst_);
    gln_apply<<<4096, 256, 0, stream>>>(t1_, gst_, P.n1w, P.n1b, skip_, kNB);   // hl -> skip_
    // global attention branch
    gemm_wmma<0><<<g192, 256, 0, stream>>>(hin, P.wq.w, P.wq.b, nullptr, qb_, kHid, kHid);
    gemm_wmma<0><<<g192, 256, 0, stream>>>(hin, P.wk.w, P.wk.b, nullptr, kb_, kHid, kHid);
    gemm_wmma<0><<<g192, 256, 0, stream>>>(hin, P.wv.w, P.wv.b, nullptr, vb_, kHid, kHid);
    if (performer)
      performer_attn<<<kNG * kHeads, 256, 0, stream>>>(qb_, kb_, vb_, P.proj, conv_);
    else
      mha_flash<<<kNG * kHeads, 256, 0, stream>>>(qb_, kb_, vb_, conv_);
    gemm_wmma<2><<<g192, 256, 0, stream>>>(conv_, P.wo.w, P.wo.b, hin, t1_, kHid, kHid);  // +h residual
    gln_stats<<<kNG, 256, 0, stream>>>(t1_, gst_);
    gln_apply<<<4096, 256, 0, stream>>>(t1_, gst_, P.n2w, P.n2b, t2_, kNB);     // hg -> t2_
    add2<<<4096, 256, 0, stream>>>(skip_, t2_, t1_, kNB);                       // out = hl + hg
    // MLP with residual
    gemm_wmma<1><<<g384, 256, 0, stream>>>(t1_, P.l1.w, P.l1.b, nullptr, mlph_, kHid, 2 * kHid);
    gemm_wmma<2><<<g192, 256, 0, stream>>>(mlph_, P.l2.w, P.l2.b, t1_, t2_, 2 * kHid, kHid);
    gln_stats<<<kNG, 256, 0, stream>>>(t2_, gst_);
    gln_apply<<<4096, 256, 0, stream>>>(t2_, gst_, P.n3w, P.n3b, dstb, kNB);
  };

  for (int L = 0; L < 2; ++L) {
    const GpsP& T = (L == 0) ? t0 : t1l;
    const GpsP& P = (L == 0) ? p0 : p1;
    const float* wcT = wcomb_ + (size_t)(2 * L) * 768;
    const float* wcP = wcomb_ + (size_t)(2 * L + 1) * 768;
    run_gps(T, false, h_, ht_, wcT);
    run_gps(P, true,  h_, hp_, wcP);
    add2<<<4096, 256, 0, stream>>>(ht_, hp_, t1_, kNB);
    hipMemsetAsync(gacc_, 0, 2 * sizeof(float), stream);
    glob_stats<<<512, 256, 0, stream>>>(t1_, gacc_, kNB);
    glob_apply<<<4096, 256, 0, stream>>>(t1_, gacc_, norm_w, norm_b, h_, kNB);
  }
  out_proj<<<kN / 256, 256, 0, stream>>>(h_, outp.w, outp.b, (float*)d_out);
}
